// SmallFourierConv_77713138254083
// MI455X (gfx1250) — matrix operand order repaired and checked by reading, not compiled, not run on hardware
//
#include <hip/hip_runtime.h>

// ---- WMMA vector types (wave32, gfx1250) ----
typedef _Float16 half8  __attribute__((ext_vector_type(8)));
typedef _Float16 half16 __attribute__((ext_vector_type(16)));
typedef float    float8 __attribute__((ext_vector_type(8)));

#define B_    8
#define HH    512
#define WW    512
#define CIN   8
#define COUT  16
#define KK    64
#define PH    576   // 512 + 64 padding (32 each side)
#define PW    576

// -------------------------------------------------------------------------
// Prep 1: f32 NHWC input -> zero-padded f16 [B][576][576][8]
// -------------------------------------------------------------------------
__global__ void pad_convert_kernel(const float* __restrict__ in,
                                   _Float16* __restrict__ xp, int total) {
  int tid = blockIdx.x * blockDim.x + threadIdx.x;
  if (tid >= total) return;
  int ci = tid & 7;
  int pw = (tid >> 3) % PW;
  int t2 = tid / (PW * 8);
  int ph = t2 % PH;
  int b  = t2 / PH;
  float v = 0.0f;
  int h = ph - 32, w = pw - 32;
  if ((unsigned)h < (unsigned)HH && (unsigned)w < (unsigned)WW)
    v = in[(((size_t)b * HH + h) * WW + w) * CIN + ci];
  xp[tid] = (_Float16)v;
}

// -------------------------------------------------------------------------
// Prep 2: restage f32 weights [64][64][8][16] into the per-lane f16
// B-fragment layout for V_WMMA_F32_16X16X32_F16:
//   wt[((kh*16 + c)*32 + lane)*16 + e]   (1 MB total, L2-resident)
//
// ISA B layout (f16, 32x16): lane L holds column n = L&15; lane group
// g = L>>4 holds K = 8g + e (e<8) and K = 16 + 8g + (e-8) (e>=8).
// Chunk reduction mapping (must match the A side): K = (kw - 4c)*8 + ci,
// so   kw = 4c + (K>>3) = 4c + g + 2*(e>=8),   ci = K&7 = e&7.
// -------------------------------------------------------------------------
__global__ void weight_restage_kernel(const float* __restrict__ ker,
                                      _Float16* __restrict__ wt) {
  int tid = blockIdx.x * blockDim.x + threadIdx.x;   // 64*16*32*16 = 524288
  int e   = tid & 15;
  int L   = (tid >> 4) & 31;
  int c   = (tid >> 9) & 15;
  int kh  = tid >> 13;
  int n   = L & 15;
  int g   = L >> 4;
  int kw  = 4 * c + g + 2 * (e >> 3);
  int ci  = e & 7;
  wt[tid] = (_Float16)ker[(((size_t)kh * KK + kw) * CIN + ci) * COUT + n];
}

// -------------------------------------------------------------------------
// Main: implicit-GEMM conv via V_WMMA_F32_16X16X32_F16.
// Block = 256 threads = 8 waves; block handles one (b,h) output row,
// each wave a 64-pixel strip (4 M-tiles of 16 px) x 16 channels.
//
// A layout (f16, 16x32; per ISA A table): lane L, m = L&15, khalf = L>>4:
//   elements 0..7  = K =      khalf*8 + j -> kw = 4c +     khalf, ci = j
//   elements 8..15 = K = 16 + khalf*8 + j -> kw = 4c + 2 + khalf, ci = j
// With NHWC-f16 padded input, each piece is one contiguous 16-byte load.
// -------------------------------------------------------------------------
__global__ __launch_bounds__(256) void conv_wmma_kernel(
    const _Float16* __restrict__ xp,   // padded f16 input
    const _Float16* __restrict__ wt,   // restaged f16 weights
    const float*    __restrict__ bias,
    float*          __restrict__ out) {
  const int bh    = blockIdx.x;        // b*512 + h
  const int b     = bh >> 9;
  const int h     = bh & 511;
  const int wave  = threadIdx.x >> 5;
  const int lane  = threadIdx.x & 31;
  const int m     = lane & 15;         // A-matrix row / output pixel in tile
  const int khalf = lane >> 4;         // which K-half this lane holds (A side)
  const int w0    = wave * 64;

  float8 acc0 = {}, acc1 = {}, acc2 = {}, acc3 = {};

  // Padded pixel for the lo A piece at chunk c: w0 + m + 63 - khalf - 4c
  const int pixbase = w0 + m + 63 - khalf;
  const half16* __restrict__ wtv = (const half16*)wt;

#pragma unroll 1
  for (int kh = 0; kh < 64; ++kh) {
    const _Float16* __restrict__ rowp =
        xp + (size_t)(b * PH + (h + 63 - kh)) * PW * CIN;
    const _Float16* __restrict__ lanep = rowp + (size_t)pixbase * CIN;
    const half16* __restrict__ wrow = wtv + (size_t)(kh * 16) * 32 + lane;
#pragma unroll 4
    for (int c = 0; c < 16; ++c) {
      // B fragment: one contiguous 32-byte load, shared by 4 WMMAs below
      half16 bfrag = wrow[c * 32];
      const _Float16* __restrict__ p = lanep - c * 32;  // -4 pixels per chunk
#pragma unroll
      for (int t = 0; t < 4; ++t) {
        const _Float16* pt = p + t * 128;               // +16 pixels per tile
        half8 lo = *(const half8*)(pt);                 // kw = 4c+khalf,   ci 0..7
        half8 hi = *(const half8*)(pt - 16);            // kw = 4c+2+khalf, ci 0..7
        half16 a = __builtin_shufflevector(lo, hi,
                    0, 1, 2, 3, 4, 5, 6, 7, 8, 9, 10, 11, 12, 13, 14, 15);
        float8 cc = (t == 0) ? acc0 : (t == 1) ? acc1 : (t == 2) ? acc2 : acc3;
        cc = __builtin_amdgcn_wmma_f32_16x16x32_f16(
                 /*neg_a=*/false, a, /*neg_b=*/false, bfrag,
                 /*c_mod=*/(short)0, cc, /*reuse_a=*/false, /*reuse_b=*/false);
        if      (t == 0) acc0 = cc;
        else if (t == 1) acc1 = cc;
        else if (t == 2) acc2 = cc;
        else             acc3 = cc;
      }
    }
  }

  // Store: 32-bit 16x16 C/D layout: lane n = lane&15, VGPR v -> M = v + 8*khalf
  const int n = m;
  const float bn = bias[n];
  const size_t obase = (size_t)(b * HH + h) * WW * COUT;
#pragma unroll
  for (int v = 0; v < 8; ++v) {
    const int mo = v + 8 * khalf;
    out[obase + (size_t)(w0 +  0 + mo) * COUT + n] = acc0[v] + bn;
    out[obase + (size_t)(w0 + 16 + mo) * COUT + n] = acc1[v] + bn;
    out[obase + (size_t)(w0 + 32 + mo) * COUT + n] = acc2[v] + bn;
    out[obase + (size_t)(w0 + 48 + mo) * COUT + n] = acc3[v] + bn;
  }
}

// -------------------------------------------------------------------------
extern "C" void kernel_launch(void* const* d_in, const int* in_sizes, int n_in,
                              void* d_out, int out_size, void* d_ws, size_t ws_size,
                              hipStream_t stream) {
  (void)in_sizes; (void)n_in; (void)out_size; (void)ws_size;
  const float* x    = (const float*)d_in[0];  // [8,512,512,8]  f32
  const float* ker  = (const float*)d_in[1];  // [64,64,8,16]   f32
  const float* bias = (const float*)d_in[2];  // [16]           f32
  float* out = (float*)d_out;                 // [8,512,512,16] f32

  _Float16* xp = (_Float16*)d_ws;                       // 42.5 MB padded input
  const size_t xp_elems = (size_t)B_ * PH * PW * CIN;   // 21,233,664 halfs
  _Float16* wt = xp + xp_elems;                         // 1 MB restaged weights

  const int pad_total = (int)xp_elems;
  pad_convert_kernel<<<(pad_total + 255) / 256, 256, 0, stream>>>(x, xp, pad_total);
  weight_restage_kernel<<<(64 * 16 * 32 * 16) / 256, 256, 0, stream>>>(ker, wt);
  conv_wmma_kernel<<<B_ * HH, 256, 0, stream>>>(xp, wt, bias, out);
}
